// Transformer_5952824673104
// MI455X (gfx1250) — matrix operand order repaired and checked by reading, not compiled, not run on hardware
//
#include <hip/hip_runtime.h>
#include <hip/hip_bf16.h>

// ---------------------------------------------------------------------------
// Types for CDNA5 WMMA (wave32, V_WMMA_F32_16X16X32_BF16)
// ---------------------------------------------------------------------------
typedef __bf16 bf16_t;
typedef __bf16 v16bf __attribute__((ext_vector_type(16)));
typedef __bf16 v8bf  __attribute__((ext_vector_type(8)));
typedef float  v8f   __attribute__((ext_vector_type(8)));
typedef unsigned int v4u __attribute__((ext_vector_type(4)));

__device__ __forceinline__ bf16_t f2bf(float f) {
    union { float f; unsigned u; } v; v.f = f;
    unsigned r = v.u + 0x7FFFu + ((v.u >> 16) & 1u);   // round-to-nearest-even
    unsigned short h = (unsigned short)(r >> 16);
    return __builtin_bit_cast(bf16_t, h);
}
__device__ __forceinline__ float bf2f(bf16_t b) {
    unsigned short h = __builtin_bit_cast(unsigned short, b);
    union { unsigned u; float f; } v; v.u = ((unsigned)h) << 16;
    return v.f;
}
__device__ __forceinline__ v8bf v8bf_zero() {
    return __builtin_bit_cast(v8bf, (v4u)0);
}

// ---------------------------------------------------------------------------
// Problem constants
// ---------------------------------------------------------------------------
constexpr int FR   = 5;          // frames
constexpr int PT   = 920;        // tokens per frame
constexpr int CD   = 512;        // d_model
constexpr int HH   = 8;          // heads
constexpr int DH   = 64;         // head dim
constexpr int BB   = 4;          // batch
constexpr int NTOK = FR * PT;    // 4600
constexpr int TROW = BB * NTOK;  // 18400 (B*N rows)
constexpr float SCALE = 0.125f;  // dh^-0.5

// ---------------------------------------------------------------------------
// Generic batched bf16 WMMA GEMM:  C[M,N] = A[M,K] @ B[K,N]  (+bias)
//   A row-major bf16 (lda).  B: BT=false -> (K,N) row-major ldb;
//   BT=true -> stored (N,K) row-major ldb (i.e. C = A @ B^T access).
//   Two-level batch: z -> (z1 = z/batch2, z2 = z%batch2).
//   Block: 128 threads = 4 waves.  Tile 64x64, BK=32.  Wave w -> rows
//   [16w,16w+16), 4 accumulators of 16x16.
//   Staging uses 16B vector chunks; requires K%8==0, and N%8==0 when
//   BT==false, plus 8-element alignment of all bases/strides (all hold here).
// ---------------------------------------------------------------------------
template<bool BT, bool OUT_BF16>
__global__ __launch_bounds__(128)
void gemm_wmma_bf16(const bf16_t* __restrict__ A, long baseA, int lda, long strideA, long strideA2,
                    const bf16_t* __restrict__ B, long baseB, int ldb, long strideB, long strideB2,
                    void* __restrict__ Cv,        long baseC, int ldc, long strideC, long strideC2,
                    const float* __restrict__ bias,
                    int M, int N, int K, int batch2)
{
    constexpr int BM = 64, BN = 64, BK = 32;
    constexpr int LDA_S = 40;   // padded LDS row stride (halfs), 80B (16B-aligned chunks)
    constexpr int LDB_S = 48;   // padded LDS row stride (halfs), 96B (32B-aligned frags)
    __shared__ alignas(64) bf16_t sA[BM][LDA_S];   // [m][k]
    __shared__ alignas(64) bf16_t sB[BN][LDB_S];   // [n][k] (k contiguous per column)

    const int tid  = threadIdx.x;
    const int wave = tid >> 5;
    const int lane = tid & 31;

    const int z  = blockIdx.z;
    const int z1 = z / batch2;
    const int z2 = z - z1 * batch2;
    const long aOff = baseA + (long)z1 * strideA + (long)z2 * strideA2;
    const long bOff = baseB + (long)z1 * strideB + (long)z2 * strideB2;
    const long cOff = baseC + (long)z1 * strideC + (long)z2 * strideC2;

    const int m0 = blockIdx.y * BM;
    const int n0 = blockIdx.x * BN;

    v8f acc[4] = {};

    const int mrow = (wave << 4) + (lane & 15);
    const int kbA  = (lane & 16) ? 16 : 0;      // A frag: 16 contiguous k per lane half
    const int kbB  = (lane & 16) ? 16 : 0;      // B frag: 16 contiguous k per lane half

    for (int k0 = 0; k0 < K; k0 += BK) {
        // ---- stage A tile: 256 x 16B chunks, 2 per thread ----
        #pragma unroll
        for (int it = 0; it < (BM * BK) / (8 * 128); ++it) {
            const int ch = tid + it * 128;
            const int mi = ch >> 2;             // 4 chunks per row
            const int kc = (ch & 3) << 3;       // 0,8,16,24
            const int gm = m0 + mi, gk = k0 + kc;
            v8bf val = v8bf_zero();
            if (gm < M && gk < K)
                val = *(const v8bf*)&A[aOff + (long)gm * lda + gk];
            *(v8bf*)&sA[mi][kc] = val;
        }
        // ---- stage B tile into [n][k] ----
        if (BT) {
            // B stored (N,K): both sides vectorized
            #pragma unroll
            for (int it = 0; it < (BN * BK) / (8 * 128); ++it) {
                const int ch = tid + it * 128;
                const int ni = ch >> 2;
                const int kc = (ch & 3) << 3;
                const int gn = n0 + ni, gk = k0 + kc;
                v8bf val = v8bf_zero();
                if (gn < N && gk < K)
                    val = *(const v8bf*)&B[bOff + (long)gn * ldb + gk];
                *(v8bf*)&sB[ni][kc] = val;
            }
        } else {
            // B stored (K,N): 16B load along n, transpose-scatter into LDS.
            // ch -> ki = ch%32 so consecutive lanes write adjacent k (banks).
            #pragma unroll
            for (int it = 0; it < (BN * BK) / (8 * 128); ++it) {
                const int ch = tid + it * 128;
                const int ki = ch & 31;
                const int nc = (ch >> 5) << 3;
                const int gk = k0 + ki, gn = n0 + nc;
                v8bf val = v8bf_zero();
                if (gk < K && gn < N)
                    val = *(const v8bf*)&B[bOff + (long)gk * ldb + gn];
                #pragma unroll
                for (int j = 0; j < 8; ++j) sB[nc + j][ki] = val[j];
            }
        }
        __syncthreads();

        // ---- fragments + 4 WMMAs ----
        const v8bf alo = *(const v8bf*)&sA[mrow][kbA];
        const v8bf ahi = *(const v8bf*)&sA[mrow][8 + kbA];
        const v16bf afrag = __builtin_shufflevector(alo, ahi,
            0, 1, 2, 3, 4, 5, 6, 7, 8, 9, 10, 11, 12, 13, 14, 15);

        #pragma unroll
        for (int nt = 0; nt < 4; ++nt) {
            const int nrow = (nt << 4) + (lane & 15);
            const v16bf bfrag = *(const v16bf*)&sB[nrow][kbB];
            acc[nt] = __builtin_amdgcn_wmma_f32_16x16x32_bf16(
                false, afrag, false, bfrag, (short)0, acc[nt], false, false);
        }
        __syncthreads();
    }

    // ---- store C (16x16 f32 layout: vgpr r -> M = r + (lane>=16 ? 8 : 0)) ----
    const int crow0 = m0 + (wave << 4) + ((lane & 16) ? 8 : 0);
    const int ccolb = n0 + (lane & 15);
    #pragma unroll
    for (int nt = 0; nt < 4; ++nt) {
        const int cc = ccolb + (nt << 4);
        if (cc >= N) continue;
        const float bv = (!OUT_BF16 && bias) ? bias[cc] : 0.0f;
        #pragma unroll
        for (int r = 0; r < 8; ++r) {
            const int rr = crow0 + r;
            if (rr < M) {
                const float v = acc[nt][r] + bv;
                if (OUT_BF16) ((bf16_t*)Cv)[cOff + (long)rr * ldc + cc] = f2bf(v);
                else          ((float*) Cv)[cOff + (long)rr * ldc + cc] = v;
            }
        }
    }
}

// ---------------------------------------------------------------------------
// Elementwise helpers
// ---------------------------------------------------------------------------
__global__ __launch_bounds__(256)
void k_f32_to_bf16(const float* __restrict__ in, bf16_t* __restrict__ out, long n)
{
    long i = (long)blockIdx.x * 256 + threadIdx.x;
    if (i < n) out[i] = f2bf(in[i]);
}

// xb[b*NTOK+n][c] = x[n][b][c]
__global__ __launch_bounds__(256)
void k_pack_x(const float* __restrict__ x, bf16_t* __restrict__ xb)
{
    long i = (long)blockIdx.x * 256 + threadIdx.x;
    if (i >= (long)TROW * CD) return;
    const int c = (int)(i & (CD - 1));
    const long t = i >> 9;               // /512
    const int b = (int)(t / NTOK);
    const int n = (int)(t - (long)b * NTOK);
    xb[i] = f2bf(x[((long)n * BB + b) * CD + c]);
}

// x_diag[t][c] = xo[t][n/PT][c]
__global__ __launch_bounds__(256)
void k_diag_gather(const bf16_t* __restrict__ xo, bf16_t* __restrict__ xd)
{
    long i = (long)blockIdx.x * 256 + threadIdx.x;
    if (i >= (long)TROW * CD) return;
    const int c = (int)(i & (CD - 1));
    const long t = i >> 9;
    const int n = (int)(t % NTOK);
    const int g = n / PT;
    xd[i] = xo[((long)t * FR + g) * (long)CD + c];
}

// d_out[(p*20 + b*5 + f)*512 + c] = final[(b*4600 + f*920 + p)*512 + c]
__global__ __launch_bounds__(256)
void k_permute_out(const float* __restrict__ fin, float* __restrict__ out)
{
    long i = (long)blockIdx.x * 256 + threadIdx.x;
    if (i >= (long)TROW * CD) return;
    const int c = (int)(i & (CD - 1));
    long r = i >> 9;
    const int m = (int)(r % (BB * FR));
    const int p = (int)(r / (BB * FR));
    const int b = m / FR, f = m - b * FR;
    out[i] = fin[((long)b * NTOK + (long)f * PT + p) * CD + c];
}

// ---------------------------------------------------------------------------
// Row softmax over P=920 with scale, fp32 -> bf16.  One block per (b,q,f).
// ---------------------------------------------------------------------------
__global__ __launch_bounds__(256)
void k_softmax_p(const float* __restrict__ scores, bf16_t* __restrict__ attn)
{
    __shared__ float smax[8];
    __shared__ float ssum[8];
    __shared__ float sbc[2];

    const int r = blockIdx.x;                 // (t*5 + f), t = b*4600+q
    const int f = r % FR;
    const long t = r / FR;
    const float* src = scores + t * (long)NTOK + (long)f * PT;
    bf16_t* dst = attn + (long)r * PT;

    const int tid = threadIdx.x, wave = tid >> 5, lane = tid & 31;

    float m = -1e30f;
    for (int p = tid; p < PT; p += 256) m = fmaxf(m, src[p] * SCALE);
    #pragma unroll
    for (int o = 16; o > 0; o >>= 1) m = fmaxf(m, __shfl_down(m, o, 32));
    if (lane == 0) smax[wave] = m;
    __syncthreads();
    if (tid == 0) {
        float w = smax[0];
        #pragma unroll
        for (int j = 1; j < 8; ++j) w = fmaxf(w, smax[j]);
        sbc[0] = w;
    }
    __syncthreads();
    const float rowmax = sbc[0];

    float s = 0.0f;
    for (int p = tid; p < PT; p += 256) s += __expf(src[p] * SCALE - rowmax);
    #pragma unroll
    for (int o = 16; o > 0; o >>= 1) s += __shfl_down(s, o, 32);
    if (lane == 0) ssum[wave] = s;
    __syncthreads();
    if (tid == 0) {
        float w = 0.0f;
        #pragma unroll
        for (int j = 0; j < 8; ++j) w += ssum[j];
        sbc[1] = 1.0f / w;
    }
    __syncthreads();
    const float inv = sbc[1];

    for (int p = tid; p < PT; p += 256)
        dst[p] = f2bf(__expf(src[p] * SCALE - rowmax) * inv);
}

// ---------------------------------------------------------------------------
// Stage-2 temporal attention: one wave per trajectory row t = b*N+n.
//   logits[h,f] = SCALE * <q2[t,h,:], k2[t,f,h,:]>, softmax over f (F=5),
//   attn2 -> d_out tail, out_pre[t,c] = sum_f a[h(c),f]*v2[t,f,c].
// Grid sized exactly: TROW/8 blocks of 256 threads (8 waves).
// ---------------------------------------------------------------------------
__global__ __launch_bounds__(256)
void k_stage2(const bf16_t* __restrict__ q2, const bf16_t* __restrict__ kv2,
              float* __restrict__ attn2, bf16_t* __restrict__ outpre)
{
    __shared__ float slog[8][48];
    const int wave = threadIdx.x >> 5, lane = threadIdx.x & 31;
    const long t = (long)blockIdx.x * 8 + wave;

    const bf16_t* qrow  = q2  + t * CD;
    const bf16_t* kvrow = kv2 + t * (long)FR * (2 * CD);

    // logits (40 = H*F pairs across 32 lanes)
    for (int p = lane; p < HH * FR; p += 32) {
        const int h = p / FR, f = p - h * FR;
        const bf16_t* qq = qrow + h * DH;
        const bf16_t* kk = kvrow + (long)f * (2 * CD) + h * DH;
        float s = 0.0f;
        #pragma unroll
        for (int d = 0; d < DH; ++d) s += bf2f(qq[d]) * bf2f(kk[d]);
        slog[wave][p] = s * SCALE;
    }
    __syncthreads();

    // per-head softmax over F=5 (lanes 0..7)
    if (lane < HH) {
        float m = -1e30f;
        #pragma unroll
        for (int f = 0; f < FR; ++f) m = fmaxf(m, slog[wave][lane * FR + f]);
        float e[FR], sum = 0.0f;
        #pragma unroll
        for (int f = 0; f < FR; ++f) { e[f] = __expf(slog[wave][lane * FR + f] - m); sum += e[f]; }
        const float inv = 1.0f / sum;
        #pragma unroll
        for (int f = 0; f < FR; ++f) slog[wave][lane * FR + f] = e[f] * inv;
    }
    __syncthreads();

    // write attn2 (B,H,N,F) fp32
    const int b = (int)(t / NTOK);
    const int n = (int)(t - (long)b * NTOK);
    for (int p = lane; p < HH * FR; p += 32) {
        const int h = p / FR, f = p - h * FR;
        attn2[(((long)(b * HH + h)) * NTOK + n) * FR + f] = slog[wave][p];
    }

    // out_pre[t, c] = sum_f a[h,f] * v2[f, c]
    for (int c = lane; c < CD; c += 32) {
        const int h = c >> 6;
        float o = 0.0f;
        #pragma unroll
        for (int f = 0; f < FR; ++f)
            o += slog[wave][h * FR + f] * bf2f(kvrow[(long)f * (2 * CD) + CD + c]);
        outpre[t * CD + c] = f2bf(o);
    }
}

// ---------------------------------------------------------------------------
// Host launch
// ---------------------------------------------------------------------------
static inline dim3 gemm_grid(int M, int N, int batches) {
    return dim3((N + 63) / 64, (M + 63) / 64, batches);
}

extern "C" void kernel_launch(void* const* d_in, const int* in_sizes, int n_in,
                              void* d_out, int out_size, void* d_ws, size_t ws_size,
                              hipStream_t stream)
{
    (void)in_sizes; (void)n_in; (void)out_size; (void)ws_size;
    const float* x      = (const float*)d_in[0];
    const float* W_qkv  = (const float*)d_in[1];
    const float* W_q2   = (const float*)d_in[2];
    const float* W_kv2  = (const float*)d_in[3];
    const float* W_proj = (const float*)d_in[4];
    const float* b_proj = (const float*)d_in[5];

    float* out_main  = (float*)d_out;                              // (P, B*F, C)
    float* out_attn2 = out_main + (long)PT * (BB * FR) * CD;       // (B, H, N, F)

    // workspace carve-up (256B aligned)
    char* ws = (char*)d_ws;
    auto alloc = [&](size_t bytes) -> char* {
        char* p = ws; ws += (bytes + 255) & ~(size_t)255; return p;
    };
    bf16_t* xb      = (bf16_t*)alloc((size_t)TROW * CD * 2);
    bf16_t* wqkv    = (bf16_t*)alloc((size_t)CD * 3 * CD * 2);
    bf16_t* wq2     = (bf16_t*)alloc((size_t)CD * CD * 2);
    bf16_t* wkv2    = (bf16_t*)alloc((size_t)CD * 2 * CD * 2);
    bf16_t* wproj   = (bf16_t*)alloc((size_t)CD * CD * 2);
    bf16_t* qkv     = (bf16_t*)alloc((size_t)TROW * 3 * CD * 2);
    float*  scores  = (float*) alloc((size_t)BB * NTOK * NTOK * 4);
    bf16_t* attn1   = (bf16_t*)alloc((size_t)BB * NTOK * FR * PT * 2);
    bf16_t* xo      = (bf16_t*)alloc((size_t)TROW * FR * CD * 2);
    bf16_t* xdiag   = (bf16_t*)alloc((size_t)TROW * CD * 2);
    bf16_t* q2b     = (bf16_t*)alloc((size_t)TROW * CD * 2);
    bf16_t* kv2b    = (bf16_t*)alloc((size_t)TROW * FR * 2 * CD * 2);
    bf16_t* outpre  = (bf16_t*)alloc((size_t)TROW * CD * 2);
    float*  finbuf  = (float*) alloc((size_t)TROW * CD * 4);

    const long EW = (long)TROW * CD;       // 9,420,800 elementwise size
    const int  EB = (int)((EW + 255) / 256);

    // ---- prep: bf16 conversions + x permute ----
    k_f32_to_bf16<<<(CD * 3 * CD + 255) / 256, 256, 0, stream>>>(W_qkv, wqkv, (long)CD * 3 * CD);
    k_f32_to_bf16<<<(CD * CD + 255) / 256, 256, 0, stream>>>(W_q2, wq2, (long)CD * CD);
    k_f32_to_bf16<<<(CD * 2 * CD + 255) / 256, 256, 0, stream>>>(W_kv2, wkv2, (long)CD * 2 * CD);
    k_f32_to_bf16<<<(CD * CD + 255) / 256, 256, 0, stream>>>(W_proj, wproj, (long)CD * CD);
    k_pack_x<<<EB, 256, 0, stream>>>(x, xb);

    // ---- GEMM1: qkv = xb @ W_qkv   (18400 x 1536, K=512) ----
    gemm_wmma_bf16<false, true><<<gemm_grid(TROW, 3 * CD, 1), 128, 0, stream>>>(
        xb, 0, CD, 0, 0,  wqkv, 0, 3 * CD, 0, 0,
        qkv, 0, 3 * CD, 0, 0,  nullptr, TROW, 3 * CD, CD, 1);

    // ---- GEMM2: scores[b] = q[b] @ k[b]^T   (4600 x 4600, K=512), batched b ----
    gemm_wmma_bf16<true, false><<<gemm_grid(NTOK, NTOK, BB), 128, 0, stream>>>(
        qkv, 0,   3 * CD, (long)NTOK * 3 * CD, 0,        // A = q (cols 0..511)
        qkv, CD,  3 * CD, (long)NTOK * 3 * CD, 0,        // B = k (cols 512..1023), transposed access
        scores, 0, NTOK, (long)NTOK * NTOK, 0,
        nullptr, NTOK, NTOK, CD, 1);

    // ---- softmax over P within each frame ----
    k_softmax_p<<<TROW * FR, 256, 0, stream>>>(scores, attn1);

    // ---- GEMM3: xo[b,:,f,:] = attn1[b,:,f,:] @ v[b,f]   (4600 x 512, K=920), batch (b,f) ----
    gemm_wmma_bf16<false, true><<<gemm_grid(NTOK, CD, BB * FR), 128, 0, stream>>>(
        attn1, 0, NTOK, (long)NTOK * FR * PT, (long)PT,             // A: lda=4600, +b*21.16M +f*920
        qkv, 2 * CD, 3 * CD, (long)NTOK * 3 * CD, (long)PT * 3 * CD,// B = v rows (b*4600+f*920 ..)
        xo, 0, FR * CD, (long)NTOK * FR * CD, (long)CD,             // C: ldc=2560, +b*11.78M +f*512
        nullptr, NTOK, CD, PT, FR);

    // ---- diagonal gather ----
    k_diag_gather<<<EB, 256, 0, stream>>>(xo, xdiag);

    // ---- GEMM q2 = x_diag @ W_q2   (18400 x 512, K=512) ----
    gemm_wmma_bf16<false, true><<<gemm_grid(TROW, CD, 1), 128, 0, stream>>>(
        xdiag, 0, CD, 0, 0,  wq2, 0, CD, 0, 0,
        q2b, 0, CD, 0, 0,  nullptr, TROW, CD, CD, 1);

    // ---- GEMM kv2 = xo_2d @ W_kv2   (92000 x 1024, K=512) ----
    gemm_wmma_bf16<false, true><<<gemm_grid(TROW * FR, 2 * CD, 1), 128, 0, stream>>>(
        xo, 0, CD, 0, 0,  wkv2, 0, 2 * CD, 0, 0,
        kv2b, 0, 2 * CD, 0, 0,  nullptr, TROW * FR, 2 * CD, CD, 1);

    // ---- stage-2 temporal attention (writes attn2 output + out_pre) ----
    k_stage2<<<TROW / 8, 256, 0, stream>>>(q2b, kv2b, out_attn2, outpre);

    // ---- GEMM5: final = out_pre @ W_proj + b_proj   (18400 x 512, K=512) ----
    gemm_wmma_bf16<false, false><<<gemm_grid(TROW, CD, 1), 128, 0, stream>>>(
        outpre, 0, CD, 0, 0,  wproj, 0, CD, 0, 0,
        finbuf, 0, CD, 0, 0,  b_proj, TROW, CD, CD, 1);

    // ---- output layout permute to (P, B*F, C) ----
    k_permute_out<<<EB, 256, 0, stream>>>(finbuf, out_main);
}
